// QuantumLSTMCell_65481071402355
// MI455X (gfx1250) — matrix operand order repaired and checked by reading, not compiled, not run on hardware
//
#include <hip/hip_runtime.h>
#include <hip/hip_bf16.h>
#include <cstdint>

#define SEQ 256
#define BATCH 512
#define IN_DIM 512
#define HID 512
#define NQ 32

typedef __attribute__((ext_vector_type(16))) __bf16 v16bf;
typedef __attribute__((ext_vector_type(8)))  float  v8f;
typedef __attribute__((ext_vector_type(8)))  unsigned int v8u;

union BF16Frag { v8u u; v16bf b; };

// ---- conversions ----------------------------------------------------------
__device__ __forceinline__ unsigned short f2bf_rne(float f) {   // weights only
  unsigned int u = __float_as_uint(f);
  u += 0x7FFFu + ((u >> 16) & 1u);
  return (unsigned short)(u >> 16);
}
__device__ __forceinline__ unsigned short f2bf_trunc(float f) {
  return (unsigned short)(__float_as_uint(f) >> 16);
}
// one v_perm_b32: two f32 -> packed {bf16(hi), bf16(lo)}
__device__ __forceinline__ unsigned int pack2bf_trunc(float lo, float hi) {
  return __builtin_amdgcn_perm(__float_as_uint(hi), __float_as_uint(lo),
                               0x07060302u);
}

// ---- fragment builders ----------------------------------------------------
// A-frag (16-bit 16x32): lane holds row M; K elems {g0: kob..+7, g1: 16+kob..+7}
__device__ __forceinline__ v16bf a_frag_from_f32(const float* g0, const float* g1) {
  float4 x0 = *(const float4*)(g0);
  float4 x1 = *(const float4*)(g0 + 4);
  float4 y0 = *(const float4*)(g1);
  float4 y1 = *(const float4*)(g1 + 4);
  BF16Frag r;
  r.u[0] = pack2bf_trunc(x0.x, x0.y);
  r.u[1] = pack2bf_trunc(x0.z, x0.w);
  r.u[2] = pack2bf_trunc(x1.x, x1.y);
  r.u[3] = pack2bf_trunc(x1.z, x1.w);
  r.u[4] = pack2bf_trunc(y0.x, y0.y);
  r.u[5] = pack2bf_trunc(y0.z, y0.w);
  r.u[6] = pack2bf_trunc(y1.x, y1.y);
  r.u[7] = pack2bf_trunc(y1.z, y1.w);
  return r.b;
}

__device__ __forceinline__ v16bf a_frag_from_lds(const unsigned short* row, int k0) {
  uint4 q0 = *(const uint4*)(row + k0);
  uint4 q1 = *(const uint4*)(row + k0 + 16);
  BF16Frag r;
  r.u[0] = q0.x; r.u[1] = q0.y; r.u[2] = q0.z; r.u[3] = q0.w;
  r.u[4] = q1.x; r.u[5] = q1.y; r.u[6] = q1.z; r.u[7] = q1.w;
  return r.b;
}

// B-frag (16-bit 32x16): lane holds column N; K elems {p..p+7, p+16..p+23} of a weight row
__device__ __forceinline__ v16bf b_frag_from_lds(const unsigned short* p) {
  uint4 q0 = *(const uint4*)(p);
  uint4 q1 = *(const uint4*)(p + 16);
  BF16Frag r;
  r.u[0] = q0.x; r.u[1] = q0.y; r.u[2] = q0.z; r.u[3] = q0.w;
  r.u[4] = q1.x; r.u[5] = q1.y; r.u[6] = q1.z; r.u[7] = q1.w;
  return r.b;
}

// ===========================================================================
// Kernel P: qx[t,b,:] = x[t,b,:] @ Wx^T + fc_b   (massively parallel WMMA)
// ===========================================================================
__global__ __launch_bounds__(256)
void qlstm_xproj_kernel(const float* __restrict__ inputs,
                        const float* __restrict__ fc_w,
                        const float* __restrict__ fc_b,
                        float* __restrict__ qx) {
  __shared__ __align__(16) unsigned short s_wx[32 * 512];  // Wx bf16, 32 KB
  __shared__ float s_fcb[32];

  for (int i = threadIdx.x; i < 32 * 512; i += 256) {
    int n = i >> 9, k = i & 511;
    s_wx[i] = f2bf_rne(fc_w[(size_t)n * 1024 + k]);
  }
  if (threadIdx.x < 32) s_fcb[threadIdx.x] = fc_b[threadIdx.x];
  __syncthreads();

  const int lane = threadIdx.x & 31;
  const int wave = threadIdx.x >> 5;
  const int nl   = lane & 15;
  const int hi   = lane >> 4;
  const int m    = lane & 15;
  const int kob  = hi * 8;
  const int kbb  = hi * 16;
  const size_t rowbase = ((size_t)blockIdx.x * 8 + wave) * 16;

  const float* xrow = inputs + (rowbase + m) * IN_DIM;
  v8f acc0, acc1;
  const float b0v = s_fcb[nl], b1v = s_fcb[16 + nl];
#pragma unroll
  for (int r = 0; r < 8; ++r) { acc0[r] = b0v; acc1[r] = b1v; }

#pragma unroll 4
  for (int kc = 0; kc < 16; ++kc) {
    v16bf a  = a_frag_from_f32(xrow + kc * 32 + kob, xrow + kc * 32 + kob + 16);
    v16bf b0 = b_frag_from_lds(s_wx + (size_t)nl * 512 + kc * 32 + kob);
    v16bf b1 = b_frag_from_lds(s_wx + (size_t)(16 + nl) * 512 + kc * 32 + kob);
    acc0 = __builtin_amdgcn_wmma_f32_16x16x32_bf16(false, a, false, b0, (short)0, acc0, false, false);
    acc1 = __builtin_amdgcn_wmma_f32_16x16x32_bf16(false, a, false, b1, (short)0, acc1, false, false);
  }

#pragma unroll
  for (int r = 0; r < 8; ++r) {
    int M = r + 8 * hi;
    qx[(rowbase + M) * NQ + nl]      = acc0[r];
    qx[(rowbase + M) * NQ + nl + 16] = acc1[r];
  }
}

// ===========================================================================
// Kernel R: sequential scan.  TWO waves per 16-row tile; h/c state in LDS.
//   Wave w owns q_in n-tile w (phase A) and out n-tiles 16w..16w+15 (phase C).
// ===========================================================================
template <bool USE_QX>
__global__ __launch_bounds__(64)
void qlstm_scan_kernel(const float* __restrict__ inputs,
                       const float* __restrict__ qx,
                       const float* __restrict__ fc_w,
                       const float* __restrict__ fc_b,
                       const float* __restrict__ out_w,
                       const float* __restrict__ out_b,
                       float* __restrict__ out) {
  __shared__ __align__(16) unsigned char smem[153728];
  unsigned short* s_wfc  = (unsigned short*)(smem);           // [32][1024] bf16
  unsigned short* s_wout = (unsigned short*)(smem + 65536);   // [512][32]  bf16
  unsigned short* s_h    = (unsigned short*)(smem + 98304);   // [16][512]  bf16
  float*          s_c    = (float*)(smem + 114688);           // [16][512]  f32
  float*          s_q    = (float*)(smem + 147456);           // [16][32]   f32
  unsigned short* s_s    = (unsigned short*)(smem + 149504);  // [16][32]   bf16
  unsigned short* s_gt   = (unsigned short*)(smem + 150528);  // [16][32]   bf16
  float*          s_fcb  = (float*)(smem + 151552);           // [32]
  float*          s_outb = (float*)(smem + 151680);           // [512]

  const int lane = threadIdx.x & 31;
  const int w    = threadIdx.x >> 5;  // wave in tile: 0 or 1
  const int nl   = lane & 15;
  const int hi   = lane >> 4;
  const int m    = lane & 15;
  const int kob  = hi * 8;
  const int kbb  = hi * 16;
  const int rowbase = blockIdx.x * 16;

  for (int i = threadIdx.x; i < 32 * 1024; i += 64) s_wfc[i]  = f2bf_rne(fc_w[i]);
  for (int i = threadIdx.x; i < 512 * 32;  i += 64) s_wout[i] = f2bf_rne(out_w[i]);
  if (threadIdx.x < 32) s_fcb[threadIdx.x] = fc_b[threadIdx.x];
  for (int i = threadIdx.x; i < 512;       i += 64) s_outb[i] = out_b[i];
  for (int i = threadIdx.x; i < 16 * 512;  i += 64) { s_h[i] = 0; s_c[i] = 0.0f; }
  __syncthreads();

  float* outputs = out;
  float* hx_out  = out + (size_t)SEQ * BATCH * HID;
  float* cx_out  = hx_out + (size_t)BATCH * HID;

  const int   ncol = w * 16 + nl;        // this wave's q_in column
  const float biasw = s_fcb[ncol];

  for (int t = 0; t < SEQ; ++t) {
    // ---- phase A: q_in[:, 16w..16w+15] = [x | h] @ fc_w^T + fc_b ----
    v8f acc;
    if (USE_QX) {
      const float* qxrow = qx + ((size_t)t * BATCH + rowbase) * NQ;
      if (t + 1 < SEQ)
        __builtin_prefetch(qx + ((size_t)(t + 1) * BATCH + rowbase) * NQ, 0, 1);
#pragma unroll
      for (int r = 0; r < 8; ++r)          // C-frag load, bias folded in pre-pass
        acc[r] = qxrow[(r + 8 * hi) * NQ + ncol];
    } else {
      const float* xrow = inputs + ((size_t)t * BATCH + rowbase + m) * IN_DIM;
      if (t + 1 < SEQ)
        __builtin_prefetch(inputs + ((size_t)(t + 1) * BATCH + rowbase + m) * IN_DIM, 0, 1);
#pragma unroll
      for (int r = 0; r < 8; ++r) acc[r] = biasw;
#pragma unroll 4
      for (int kc = 0; kc < 16; ++kc) {    // x half, K = 0..511
        v16bf a = a_frag_from_f32(xrow + kc * 32 + kob, xrow + kc * 32 + kob + 16);
        v16bf b = b_frag_from_lds(s_wfc + (size_t)ncol * 1024 + kc * 32 + kob);
        acc = __builtin_amdgcn_wmma_f32_16x16x32_bf16(false, a, false, b, (short)0, acc, false, false);
      }
    }
#pragma unroll 4
    for (int kc = 0; kc < 16; ++kc) {      // h half, K = 512..1023 (LDS)
      v16bf a = a_frag_from_lds(s_h + (size_t)m * 512, kc * 32 + kob);
      v16bf b = b_frag_from_lds(s_wfc + (size_t)ncol * 1024 + 512 + kc * 32 + kob);
      acc = __builtin_amdgcn_wmma_f32_16x16x32_bf16(false, a, false, b, (short)0, acc, false, false);
    }

    // scatter q_in C-frag to LDS
#pragma unroll
    for (int r = 0; r < 8; ++r)
      s_q[(r + 8 * hi) * 32 + ncol] = acc[r];
    __syncthreads();

    // ---- phase B: q = cumprod(cos(q_in)) per row; activations to bf16 ----
    if (threadIdx.x < 16) {
      float prod = 1.0f;
#pragma unroll 4
      for (int j = 0; j < NQ; ++j) {
        prod *= __cosf(s_q[m * 32 + j]);
        float sg = 1.0f / (1.0f + __expf(-prod));
        s_s[m * 32 + j]  = f2bf_trunc(sg);
        s_gt[m * 32 + j] = f2bf_trunc(tanhf(prod));
      }
    }
    __syncthreads();

    // ---- phase C: f/g = act(q) @ out_w^T + out_b; cell update ----
    v16bf sa = a_frag_from_lds(s_s + m * 32, kob);
    v16bf ga = a_frag_from_lds(s_gt + m * 32, kob);

#pragma unroll 4
    for (int i = 0; i < 16; ++i) {
      int nt = w * 16 + i;                 // this wave's out n-tiles
      int n  = nt * 16 + nl;
      v16bf bw = b_frag_from_lds(s_wout + (size_t)n * 32 + kob);
      float bv = s_outb[n];
      v8f cf, cg;
#pragma unroll
      for (int r = 0; r < 8; ++r) { cf[r] = bv; cg[r] = bv; }
      cf = __builtin_amdgcn_wmma_f32_16x16x32_bf16(false, sa, false, bw, (short)0, cf, false, false);
      cg = __builtin_amdgcn_wmma_f32_16x16x32_bf16(false, ga, false, bw, (short)0, cg, false, false);
#pragma unroll
      for (int r = 0; r < 8; ++r) {
        int M = r + 8 * hi;
        float fv = cf[r];
        float gv = cg[r];
        float cold = s_c[M * 512 + n];
        float cn = fv * (cold + gv);          // c = f*c + i*g, i == f
        s_c[M * 512 + n] = cn;
        float hv = fv * tanhf(cn);            // h = o*tanh(c), o == f
        s_h[M * 512 + n] = f2bf_trunc(hv);
        outputs[((size_t)t * BATCH + rowbase + M) * HID + n] = hv;
        if (t == SEQ - 1) {
          hx_out[(size_t)(rowbase + M) * HID + n] = hv;
          cx_out[(size_t)(rowbase + M) * HID + n] = cn;
        }
      }
    }
    __syncthreads();
  }
}

extern "C" void kernel_launch(void* const* d_in, const int* in_sizes, int n_in,
                              void* d_out, int out_size, void* d_ws, size_t ws_size,
                              hipStream_t stream) {
  const float* inputs = (const float*)d_in[0];
  const float* fc_w   = (const float*)d_in[1];
  const float* fc_b   = (const float*)d_in[2];
  const float* out_w  = (const float*)d_in[3];
  const float* out_b  = (const float*)d_in[4];
  float* out = (float*)d_out;
  (void)in_sizes; (void)n_in; (void)out_size;

  const size_t qx_bytes = (size_t)SEQ * BATCH * NQ * sizeof(float);
  if (d_ws != nullptr && ws_size >= qx_bytes) {
    float* qx = (float*)d_ws;
    qlstm_xproj_kernel<<<dim3((SEQ * BATCH / 16) / 8), dim3(256), 0, stream>>>(
        inputs, fc_w, fc_b, qx);
    qlstm_scan_kernel<true><<<dim3(BATCH / 16), dim3(64), 0, stream>>>(
        inputs, qx, fc_w, fc_b, out_w, out_b, out);
  } else {
    qlstm_scan_kernel<false><<<dim3(BATCH / 16), dim3(64), 0, stream>>>(
        inputs, nullptr, fc_w, fc_b, out_w, out_b, out);
  }
}
